// NonLocalAttentionBlock_28209345200524
// MI455X (gfx1250) — matrix operand order repaired and checked by reading, not compiled, not run on hardware
//
#include <hip/hip_runtime.h>
#include <hip/hip_bf16.h>
#include <math.h>

// ---------------------------------------------------------------------------
// MI455X (gfx1250) implementation of the NonLocalAttentionBlock reference.
// All matmuls (1x1 convs + window channel attention) use v_wmma_f32_16x16x32_f16.
// GEMM A-tiles are staged with GLOBAL_LOAD_ASYNC_TO_LDS_B128 when available.
// Intermediates are pixel-major (NHWC) f16 staged through d_ws (lives in 192MB L2).
// ---------------------------------------------------------------------------

typedef __attribute__((ext_vector_type(8)))  _Float16 v8h;
typedef __attribute__((ext_vector_type(16))) _Float16 v16h;
typedef __attribute__((ext_vector_type(8)))  float    v8f;
typedef __attribute__((ext_vector_type(4)))  int      v4i;

#define NAB_B   2
#define NAB_C   192
#define NAB_H   256
#define NAB_W   256
#define NAB_P   (NAB_B * NAB_H * NAB_W)   // 131072 pixels
#define NAB_HEADS 6
#define NAB_CH  32                        // channels per head
#define NAB_WS  8                         // window size
#define NAB_HID 384

#if defined(__gfx1250__) && __has_builtin(__builtin_amdgcn_global_load_async_to_lds_b128)
#define NAB_ASYNC_LDS 1
#else
#define NAB_ASYNC_LDS 0
#endif

// 16-byte global -> LDS stage (async DMA on gfx1250, sync fallback otherwise).
static __device__ __forceinline__ void stage16(const _Float16* src, _Float16* dst) {
#if NAB_ASYNC_LDS
  typedef __attribute__((address_space(1))) v4i* gp_t;
  typedef __attribute__((address_space(3))) v4i* lp_t;
  __builtin_amdgcn_global_load_async_to_lds_b128(
      (gp_t)(uintptr_t)src, (lp_t)(unsigned int)(uintptr_t)dst, 0, 0);
#else
  *(v8h*)dst = *(const v8h*)src;
#endif
}

static __device__ __forceinline__ void wait_async_stage() {
#if NAB_ASYNC_LDS
  asm volatile("s_wait_asynccnt 0" ::: "memory");
#endif
}

static __device__ __forceinline__ v16h cat16(v8h lo, v8h hi) {
  return __builtin_shufflevector(lo, hi, 0,1,2,3,4,5,6,7,8,9,10,11,12,13,14,15);
}

// A-fragment (16x32 f16, row-major LDS, row stride S halves).
// Layout: M = lane%16 ; K = (h>=8?16:0) + (lane>=16?8:0) + (h%8)
static __device__ __forceinline__ v16h load_a_frag(const _Float16* base, int S,
                                                   int m0, int k0, int lane) {
  const int m  = m0 + (lane & 15);
  const int kb = k0 + ((lane >> 4) ? 8 : 0);
  v8h lo = *(const v8h*)(base + m * S + kb);
  v8h hi = *(const v8h*)(base + m * S + kb + 16);
  return cat16(lo, hi);
}

// B-fragment (32x16 f16). LDS holds B transposed: n-major, row stride S halves,
// i.e. base[n*S + k] = B[k][n].  Layout: N = lane%16 ; K = (h>=8?16:0) + (lane>=16?8:0) + (h%8)
static __device__ __forceinline__ v16h load_b_frag(const _Float16* base, int S,
                                                   int n0, int k0, int lane) {
  const int n  = n0 + (lane & 15);
  const int kb = k0 + ((lane >> 4) ? 8 : 0);
  v8h lo = *(const v8h*)(base + n * S + kb);
  v8h hi = *(const v8h*)(base + n * S + kb + 16);
  return cat16(lo, hi);
}

static __device__ __forceinline__ float gelu_exact(float x) {
  return 0.5f * x * (1.0f + erff(x * 0.70710678118654752f));
}

// ---------------------------------------------------------------------------
// LayerNorm over channels (192) per pixel.  NCHW=true reads [b,c,h,w] fp32,
// else reads pixel-major [p,c] fp32.  Writes pixel-major f16 [p,192].
// ---------------------------------------------------------------------------
template <bool NCHW>
__global__ __launch_bounds__(256) void ln_kernel(const float* __restrict__ x,
                                                 const float* __restrict__ w,
                                                 const float* __restrict__ b,
                                                 _Float16* __restrict__ out) {
  const int p  = blockIdx.x * 256 + threadIdx.x;
  const int bi = p >> 16;
  const int yx = p & 65535;
  const float* src;
  int stride;
  if (NCHW) { src = x + (size_t)bi * NAB_C * 65536 + yx; stride = 65536; }
  else      { src = x + (size_t)p * NAB_C;               stride = 1; }

  float s = 0.f, s2 = 0.f;
  for (int c = 0; c < NAB_C; ++c) {
    float v = src[(size_t)c * stride];
    s += v; s2 += v * v;
  }
  const float mu   = s * (1.0f / NAB_C);
  const float var  = s2 * (1.0f / NAB_C) - mu * mu;
  const float rstd = rsqrtf(var + 1e-5f);
  _Float16* dst = out + (size_t)p * NAB_C;
  for (int c = 0; c < NAB_C; ++c) {
    float v = src[(size_t)c * stride];
    dst[c] = (_Float16)((v - mu) * rstd * w[c] + b[c]);
  }
}

// ---------------------------------------------------------------------------
// WMMA GEMM: out[M,N] = A[M,K] (f16 pixel-major) x W[N,K]^T (f32 weights).
// Block tile 256x64, BK=32, 256 threads = 8 waves, two 16-row strips per wave
// (B-fragments reused across both strips).  A staged via async LDS DMA.
// MODE 0: store f16 pixel-major.
// MODE 1: += residual read NCHW f32, store f32 pixel-major (N == 192).
// MODE 2: += residual read pixel-major f32, store f32 NCHW (N == 192).
// ---------------------------------------------------------------------------
template <int MODE>
__global__ __launch_bounds__(256) void gemm_kernel(const _Float16* __restrict__ A,
                                                   const float* __restrict__ W,
                                                   const float* __restrict__ R,
                                                   _Float16* __restrict__ OH,
                                                   float* __restrict__ OF,
                                                   int N, int K) {
  constexpr int BM = 256, BN = 64, BK = 32;
  __shared__ __align__(16) _Float16 As[BM * BK];  // 16 KB, m-major [256][32]
  __shared__ __align__(16) _Float16 Bs[BN * BK];  //  4 KB, n-major [64][32]

  const int tid  = threadIdx.x;
  const int lane = tid & 31;
  const int wave = tid >> 5;
  const int m0   = blockIdx.x * BM;
  const int n0b  = blockIdx.y * BN;

  v8f acc[2][4] = {};

  for (int k0 = 0; k0 < K; k0 += BK) {
    // --- stage A tile: 256 rows x 32 halves = 1024 b128 DMAs, 4/thread ---
    #pragma unroll
    for (int i = 0; i < 4; ++i) {
      int idx = tid + i * 256;             // 0..1023
      int r = idx >> 2, v = idx & 3;
      stage16(A + (size_t)(m0 + r) * K + k0 + v * 8, As + r * BK + v * 8);
    }
    // --- stage B tile (transposed, f32->f16): 64 rows x 32, 4 threads/row ---
    {
      int r = tid >> 2, j = (tid & 3) * 8;
      const float* wp = W + (size_t)(n0b + r) * K + k0 + j;
      v8h t;
      #pragma unroll
      for (int e = 0; e < 8; ++e) t[e] = (_Float16)wp[e];
      *(v8h*)(Bs + r * BK + j) = t;
      if (k0 + BK < K) __builtin_prefetch(wp + BK, 0, 1);
    }
    wait_async_stage();
    __syncthreads();

    const v16h a0 = load_a_frag(As, BK, wave * 32, 0, lane);
    const v16h a1 = load_a_frag(As, BK, wave * 32 + 16, 0, lane);
    #pragma unroll
    for (int nt = 0; nt < 4; ++nt) {
      const v16h b = load_b_frag(Bs, BK, nt * 16, 0, lane);
      acc[0][nt] = __builtin_amdgcn_wmma_f32_16x16x32_f16(
          false, a0, false, b, (short)0, acc[0][nt], false, false);
      acc[1][nt] = __builtin_amdgcn_wmma_f32_16x16x32_f16(
          false, a1, false, b, (short)0, acc[1][nt], false, false);
    }
    __syncthreads();
  }

  // --- epilogue.  C layout: M = r + (lane>=16?8:0), N = lane%16 ---
  const int hi8  = (lane >> 4) ? 8 : 0;
  const int ncol = lane & 15;
  #pragma unroll
  for (int s = 0; s < 2; ++s) {
    #pragma unroll
    for (int nt = 0; nt < 4; ++nt) {
      #pragma unroll
      for (int r = 0; r < 8; ++r) {
        const int gm = m0 + wave * 32 + s * 16 + r + hi8;   // pixel
        const int gn = n0b + nt * 16 + ncol;                // out channel
        const float v = acc[s][nt][r];
        if (MODE == 0) {
          OH[(size_t)gm * N + gn] = (_Float16)v;
        } else if (MODE == 1) {
          const int bi = gm >> 16, yx = gm & 65535;
          const float res = R[((size_t)bi * NAB_C + gn) * 65536 + yx];
          OF[(size_t)gm * NAB_C + gn] = v + res;
        } else {
          const int bi = gm >> 16, yx = gm & 65535;
          const float res = R[(size_t)gm * NAB_C + gn];
          OF[((size_t)bi * NAB_C + gn) * 65536 + yx] = v + res;
        }
      }
    }
  }
}

// ---------------------------------------------------------------------------
// Depthwise 3x3 "SAME" on pixel-major f16 [P, C], 8 channels per thread
// (b128 loads/stores).  Weights f32 [C,3,3].
// ---------------------------------------------------------------------------
__global__ __launch_bounds__(256) void dwconv8_kernel(const _Float16* __restrict__ in,
                                                      const float* __restrict__ w,
                                                      _Float16* __restrict__ out,
                                                      int C) {
  const int G = C >> 3;                       // channel groups of 8
  const size_t gid = (size_t)blockIdx.x * 256 + threadIdx.x;
  const int g = (int)(gid % G);
  const size_t p = gid / G;
  const int bi = (int)(p >> 16), yx = (int)(p & 65535);
  const int y = yx >> 8, x = yx & 255;
  const int c0 = g * 8;

  float acc[8] = {};
  #pragma unroll
  for (int dy = -1; dy <= 1; ++dy) {
    const int yy = y + dy;
    if (yy < 0 || yy > 255) continue;
    #pragma unroll
    for (int dx = -1; dx <= 1; ++dx) {
      const int xx = x + dx;
      if (xx < 0 || xx > 255) continue;
      const size_t pp = ((size_t)bi << 16) + yy * 256 + xx;
      const v8h t = *(const v8h*)(in + pp * C + c0);
      const int tap = (dy + 1) * 3 + (dx + 1);
      #pragma unroll
      for (int e = 0; e < 8; ++e)
        acc[e] += (float)t[e] * w[(c0 + e) * 9 + tap];
    }
  }
  v8h o;
  #pragma unroll
  for (int e = 0; e < 8; ++e) o[e] = (_Float16)acc[e];
  *(v8h*)(out + p * C + c0) = o;
}

// ---------------------------------------------------------------------------
// Fused dw3x3 on [P,768] + GDFN gate: out[p,c<384] = gelu(dw(c)) * dw(c+384),
// 8 channels per thread.
// ---------------------------------------------------------------------------
__global__ __launch_bounds__(256) void dwgate8_kernel(const _Float16* __restrict__ in,
                                                      const float* __restrict__ w,
                                                      _Float16* __restrict__ out) {
  constexpr int C2 = 2 * NAB_HID;             // 768
  constexpr int G = NAB_HID / 8;              // 48 groups
  const size_t gid = (size_t)blockIdx.x * 256 + threadIdx.x;
  const int g = (int)(gid % G);
  const size_t p = gid / G;
  const int bi = (int)(p >> 16), yx = (int)(p & 65535);
  const int y = yx >> 8, x = yx & 255;
  const int c0 = g * 8;

  float a1[8] = {}, a2[8] = {};
  #pragma unroll
  for (int dy = -1; dy <= 1; ++dy) {
    const int yy = y + dy;
    if (yy < 0 || yy > 255) continue;
    #pragma unroll
    for (int dx = -1; dx <= 1; ++dx) {
      const int xx = x + dx;
      if (xx < 0 || xx > 255) continue;
      const size_t pp = ((size_t)bi << 16) + yy * 256 + xx;
      const _Float16* s = in + pp * C2;
      const v8h t1 = *(const v8h*)(s + c0);
      const v8h t2 = *(const v8h*)(s + NAB_HID + c0);
      const int tap = (dy + 1) * 3 + (dx + 1);
      #pragma unroll
      for (int e = 0; e < 8; ++e) {
        a1[e] += (float)t1[e] * w[(c0 + e) * 9 + tap];
        a2[e] += (float)t2[e] * w[(NAB_HID + c0 + e) * 9 + tap];
      }
    }
  }
  v8h o;
  #pragma unroll
  for (int e = 0; e < 8; ++e) o[e] = (_Float16)(gelu_exact(a1[e]) * a2[e]);
  *(v8h*)(out + p * NAB_HID + c0) = o;
}

// ---------------------------------------------------------------------------
// Windowed channel attention.  One wave per (window, head):
//   q,k,v : 32ch x 64pos tiles from dw(qkv) [P,576] f16
//   l2-normalize q,k rows; attn = relu(temp * q k^T) [32x32, K=64]
//   out   = attn v [32x64, K=32];  write f16 pixel-major [p,192].
// WMMA 16x16x32, LDS-staged, 4 waves / block.
// ---------------------------------------------------------------------------
__global__ __launch_bounds__(128) void attention_kernel(const _Float16* __restrict__ dwq,
                                                        const float* __restrict__ temp,
                                                        _Float16* __restrict__ out) {
  constexpr int QS = 32 * 64, KS = 32 * 64, VT = 64 * 32, AS = 32 * 32;
  constexpr int PW = QS + KS + VT + AS;             // 7168 halves / wave
  __shared__ __align__(16) _Float16 smem[4 * PW];   // 56 KB

  const int lane = threadIdx.x & 31;
  const int wave = threadIdx.x >> 5;
  const int unit = blockIdx.x * 4 + wave;           // 12288 units
  const int head = unit % NAB_HEADS;
  const int win  = unit / NAB_HEADS;
  const int bi = win >> 10, rem = win & 1023;
  const int wy = rem >> 5, wx = rem & 31;
  const int pbase = bi * 65536 + (wy * NAB_WS) * 256 + wx * NAB_WS;

  _Float16* q_s = smem + wave * PW;     // [32][64] c-major
  _Float16* k_s = q_s + QS;             // [32][64]
  _Float16* v_t = k_s + KS;             // [64][32] pos-major (B-transposed)
  _Float16* a_s = v_t + VT;             // [32][32] attn, c-major

  // ---- load q/k/v for this wave's (window, head); lane = channel ----
  const int coff = head * NAB_CH + lane;
  float sq = 0.f, sk = 0.f;
  for (int pos = 0; pos < 64; ++pos) {
    const int p = pbase + (pos >> 3) * 256 + (pos & 7);
    const _Float16* s = dwq + (size_t)p * (3 * NAB_C) + coff;
    const _Float16 qv = s[0], kv = s[NAB_C], vv = s[2 * NAB_C];
    q_s[lane * 64 + pos] = qv;
    k_s[lane * 64 + pos] = kv;
    v_t[pos * 32 + lane] = vv;
    sq += (float)qv * (float)qv;
    sk += (float)kv * (float)kv;
  }
  const float qsc = 1.0f / fmaxf(sqrtf(sq), 1e-12f);
  const float ksc = 1.0f / fmaxf(sqrtf(sk), 1e-12f);
  for (int pos = 0; pos < 64; ++pos) {
    q_s[lane * 64 + pos] = (_Float16)((float)q_s[lane * 64 + pos] * qsc);
    k_s[lane * 64 + pos] = (_Float16)((float)k_s[lane * 64 + pos] * ksc);
  }
  __syncthreads();

  const float tv   = temp[head];
  const int   hi8  = (lane >> 4) ? 8 : 0;
  const int   ncol = lane & 15;

  // ---- attn = q k^T : M=32 N=32 K=64 ;  B[k][n] = k_s[n][k] (n-major)
  #pragma unroll
  for (int mt = 0; mt < 32; mt += 16) {
    #pragma unroll
    for (int nt = 0; nt < 32; nt += 16) {
      v8f acc = {};
      #pragma unroll
      for (int kc = 0; kc < 64; kc += 32) {
        const v16h a = load_a_frag(q_s, 64, mt, kc, lane);
        const v16h b = load_b_frag(k_s, 64, nt, kc, lane);
        acc = __builtin_amdgcn_wmma_f32_16x16x32_f16(
            false, a, false, b, (short)0, acc, false, false);
      }
      #pragma unroll
      for (int r = 0; r < 8; ++r) {
        const int m = mt + r + hi8;
        const int n = nt + ncol;
        a_s[m * 32 + n] = (_Float16)fmaxf(acc[r] * tv, 0.0f);
      }
    }
  }
  __syncthreads();

  // ---- out = attn v : M=32 N=64 K=32 ;  B[k][n] = v_t[n][k] (pos-major)
  #pragma unroll
  for (int mt = 0; mt < 32; mt += 16) {
    const v16h a = load_a_frag(a_s, 32, mt, 0, lane);
    #pragma unroll
    for (int nt = 0; nt < 64; nt += 16) {
      const v16h b = load_b_frag(v_t, 32, nt, 0, lane);
      v8f acc = {};
      acc = __builtin_amdgcn_wmma_f32_16x16x32_f16(
          false, a, false, b, (short)0, acc, false, false);
      #pragma unroll
      for (int r = 0; r < 8; ++r) {
        const int m   = mt + r + hi8;          // channel within head
        const int pos = nt + ncol;
        const int p   = pbase + (pos >> 3) * 256 + (pos & 7);
        out[(size_t)p * NAB_C + head * NAB_CH + m] = (_Float16)acc[r];
      }
    }
  }
}

// ---------------------------------------------------------------------------
extern "C" void kernel_launch(void* const* d_in, const int* in_sizes, int n_in,
                              void* d_out, int out_size, void* d_ws, size_t ws_size,
                              hipStream_t stream) {
  (void)in_sizes; (void)n_in; (void)out_size; (void)ws_size;

  const float* x      = (const float*)d_in[0];
  const float* ln3_w  = (const float*)d_in[1];
  const float* ln3_b  = (const float*)d_in[2];
  const float* temp   = (const float*)d_in[3];
  const float* W_qkv  = (const float*)d_in[4];   // [576,192]
  const float* W_qdw  = (const float*)d_in[5];   // [576,3,3]
  const float* W_po   = (const float*)d_in[6];   // [192,192]
  const float* ln4_w  = (const float*)d_in[7];
  const float* ln4_b  = (const float*)d_in[8];
  const float* W_pi   = (const float*)d_in[9];   // [768,192]
  const float* W_fdw  = (const float*)d_in[10];  // [768,3,3]
  const float* W_fo   = (const float*)d_in[11];  // [192,384]
  float* out = (float*)d_out;

  // workspace carve-up
  char* ws = (char*)d_ws;
  const size_t szXN = (size_t)NAB_P * NAB_C * 2;        //  48 MB  f16 [P,192]
  const size_t szA  = (size_t)NAB_P * 768 * 2;          // 192 MB  f16 [P,<=768]
  const size_t szB  = (size_t)NAB_P * 768 * 2;          // 192 MB  f16 [P,<=768]
  _Float16* xn   = (_Float16*)(ws);                     // LN outputs
  _Float16* bufA = (_Float16*)(ws + szXN);              // qkv / attn_out / pi
  _Float16* bufB = (_Float16*)(ws + szXN + szA);        // dw(qkv) / gated
  float*    x1   = (float*)   (ws + szXN + szA + szB);  // f32 [P,192] pixel-major

  const int P = NAB_P;

  // 1) LN3 (NCHW in) -> xn f16 [P,192]
  ln_kernel<true><<<P / 256, 256, 0, stream>>>(x, ln3_w, ln3_b, xn);
  // 2) qkv = xn @ W_qkv^T  -> bufA f16 [P,576]
  gemm_kernel<0><<<dim3(P / 256, 576 / 64), 256, 0, stream>>>(
      xn, W_qkv, nullptr, bufA, nullptr, 576, NAB_C);
  // 3) depthwise 3x3 on 576 ch -> bufB
  dwconv8_kernel<<<(size_t)P * (576 / 8) / 256, 256, 0, stream>>>(bufA, W_qdw, bufB, 576);
  // 4) window channel attention -> bufA f16 [P,192]
  attention_kernel<<<(P / 64) * NAB_HEADS / 4, 128, 0, stream>>>(bufB, temp, bufA);
  // 5) x1 = attn_out @ W_po^T + x  -> x1 f32 [P,192] pixel-major
  gemm_kernel<1><<<dim3(P / 256, 192 / 64), 256, 0, stream>>>(
      bufA, W_po, x, nullptr, x1, NAB_C, NAB_C);
  // 6) LN4 (pixel-major in) -> xn
  ln_kernel<false><<<P / 256, 256, 0, stream>>>(x1, ln4_w, ln4_b, xn);
  // 7) pi = xn @ W_pi^T -> bufA f16 [P,768]
  gemm_kernel<0><<<dim3(P / 256, 768 / 64), 256, 0, stream>>>(
      xn, W_pi, nullptr, bufA, nullptr, 768, NAB_C);
  // 8) fused dw3x3(768) + gelu-gate -> bufB f16 [P,384]
  dwgate8_kernel<<<(size_t)P * (NAB_HID / 8) / 256, 256, 0, stream>>>(bufA, W_fdw, bufB);
  // 9) out = gated @ W_fo^T + x1 -> NCHW f32
  gemm_kernel<2><<<dim3(P / 256, 192 / 64), 256, 0, stream>>>(
      bufB, W_fo, x1, nullptr, out, NAB_C, NAB_HID);
}
